// VLSTMNoisy_73435350827526
// MI455X (gfx1250) — matrix operand order repaired and checked by reading, not compiled, not run on hardware
//
#include <hip/hip_runtime.h>
#include <hip/hip_bf16.h>

// ---------------------------------------------------------------------------
// VLSTMNoisy on MI455X (gfx1250).
// BS=256, OBS=50, PRED=25, Tn=49, T=74, F=96, H=1024, G=4H=4096.
//
//  * One-time per call: convert all LSTM/FC weights fp32 -> bf16, pre-swizzled
//    into WMMA B-fragment blocks (512 bf16 per 32x16 K-chunk). ~78MB total,
//    fits the 192MB L2 -> per-step weight streaming never touches HBM.
//  * Per timestep, per LSTM layer: one fused kernel. WG = 256 threads
//    (8 waves), tile = 64 rows (batch) x 64 cols (hidden). Each wave owns a
//    16-col strip and 32 rows -> 2 M-frags x 4 gates = 8 v_wmma accumulators.
//    K loop uses pure pointer-increment addressing and 2x unroll so loads of
//    chunk k+1 overlap the WMMAs of chunk k. LSTM nonlinearity fused
//    in-register; c updated in place (fp32), h written bf16 (ping-pong).
//  * FC head (96x1024) is a 1-wave-per-16x16-tile WMMA kernel fused with the
//    clip and the bf16 autoregressive feedback write.
// ---------------------------------------------------------------------------

typedef unsigned short u16;                                  // raw bf16 bits
typedef __attribute__((ext_vector_type(16))) __bf16 v16bf;
typedef __attribute__((ext_vector_type(8)))  float  v8f;

union Frag { uint4 q[2]; v16bf v; };

__device__ __forceinline__ uint4 ld_uint4(const u16* p) {
  return *(const uint4*)p;
}

__device__ __forceinline__ u16 f2bf(float f) {
  unsigned u = __float_as_uint(f);
  u += 0x7FFFu + ((u >> 16) & 1u);   // round-to-nearest-even
  return (u16)(u >> 16);
}

__device__ __forceinline__ float sigf(float x) {
  return 1.0f / (1.0f + __expf(-x));
}

// ---------------------------------------------------------------------------
// Weight pack: row-major W[N][K] fp32 -> bf16 WMMA-B fragment blocks.
// Block (n16,k32) holds a 32(K)x16(N) tile: lane l owns N=n16*16+(l&15),
// K = k32*32 + ((l>=16)?16:0) + e, e=0..15, stored contiguously (32B/lane).
// ---------------------------------------------------------------------------
__global__ void pack_w_kernel(const float* __restrict__ W, u16* __restrict__ P,
                              int N, int K) {
  int tid = blockIdx.x * blockDim.x + threadIdx.x;
  int K32 = K >> 5;
  int total = (N >> 4) * K32 * 32;       // (block, lane) pairs
  if (tid >= total) return;
  int lane = tid & 31;
  int blk  = tid >> 5;
  int n16  = blk / K32;
  int k32  = blk % K32;
  int n    = n16 * 16 + (lane & 15);
  int kb   = k32 * 32 + ((lane >= 16) ? 16 : 0);
  const float* src = W + (size_t)n * K + kb;
  u16* dst = P + (size_t)tid * 16;
#pragma unroll
  for (int e = 0; e < 16; ++e) dst[e] = f2bf(src[e]);
}

__global__ void bias_kernel(const float* __restrict__ a,
                            const float* __restrict__ b,
                            float* __restrict__ o, int n) {
  int t = blockIdx.x * blockDim.x + threadIdx.x;
  if (t < n) o[t] = a[t] + b[t];
}

// ---------------------------------------------------------------------------
// Build noisy-velocity input sequence X[t][m][k] (bf16), t in [0,74).
// ---------------------------------------------------------------------------
__global__ void prep_x_kernel(const float* __restrict__ pose,
                              const int* __restrict__ noise,
                              u16* __restrict__ X) {
  int tid = blockIdx.x * blockDim.x + threadIdx.x;
  const int total = 74 * 256 * 96;
  if (tid >= total) return;
  int t = tid / (256 * 96);
  int rem = tid % (256 * 96);
  int m = rem / 96, k = rem % 96;
  float v = 0.0f;
  if (t < 49) {
    v = pose[(size_t)m * 4800 + (t + 1) * 96 + k] -
        pose[(size_t)m * 4800 + t * 96 + k];
    int nz = noise[(size_t)m * 1600 + (t + 1) * 32 + k / 3];
    if (nz == 1) v = 0.0f;
  }
  X[(size_t)t * 24576 + m * 96 + k] = f2bf(v);
}

// ---------------------------------------------------------------------------
// Fused LSTM cell step.  grid = (4,16)  block = 256 (8 waves).
// gates = X @ Wx^T + Hin @ Wh^T + bias;  c,h update fused.
// ---------------------------------------------------------------------------
__global__ __launch_bounds__(256) void lstm_cell_kernel(
    const u16* __restrict__ X, int ldx, int Kx,
    const u16* __restrict__ Hin,
    const u16* __restrict__ Wx, const u16* __restrict__ Wh,
    const float* __restrict__ bias,
    float* __restrict__ C, u16* __restrict__ Hout) {
  const int lane = threadIdx.x & 31;
  const int wv   = threadIdx.x >> 5;
  const int wc   = wv & 3;    // which 16-col strip of the 64-col tile
  const int wm   = wv >> 2;   // which 32-row half of the 64-row tile
  const int mtile = blockIdx.x * 64;
  const int ntile = blockIdx.y * 64;
  const int ncol  = ntile + wc * 16 + (lane & 15);    // N for B/C/D frags
  const int kbaseA = (lane >= 16) ? 16 : 0;

  v8f acc[2][4];
#pragma unroll
  for (int fm = 0; fm < 2; ++fm)
#pragma unroll
    for (int g = 0; g < 4; ++g) {
      float b = bias[g * 1024 + ncol];
#pragma unroll
      for (int r = 0; r < 8; ++r) acc[fm][g][r] = b;
    }

  const int rowA0 = mtile + wm * 32 + (lane & 15);

  for (int phase = 0; phase < 2; ++phase) {
    const u16* __restrict__ A   = phase ? Hin : X;
    const int               lda = phase ? 1024 : ldx;
    const int               K   = phase ? 1024 : Kx;
    const u16* __restrict__ W   = phase ? Wh : Wx;
    const int               K32 = K >> 5;

    // Pointer-increment addressing: A rows advance 32 bf16 (64B) per chunk,
    // B fragment blocks advance 512 bf16 (1KB) per chunk.
    const u16* ap0 = A + (size_t)rowA0 * lda + kbaseA;
    const u16* ap1 = A + (size_t)(rowA0 + 16) * lda + kbaseA;
    const u16* bp0;
    const u16* bp1;
    const u16* bp2;
    const u16* bp3;
    {
      const int nb = (ntile >> 4) + wc;
      bp0 = W + ((size_t)((0 * 64 + nb) * K32) * 32 + lane) * 16;
      bp1 = W + ((size_t)((1 * 64 + nb) * K32) * 32 + lane) * 16;
      bp2 = W + ((size_t)((2 * 64 + nb) * K32) * 32 + lane) * 16;
      bp3 = W + ((size_t)((3 * 64 + nb) * K32) * 32 + lane) * 16;
    }

#pragma unroll 2
    for (int k32 = 0; k32 < K32; ++k32) {
      Frag af[2];
      af[0].q[0] = ld_uint4(ap0);
      af[0].q[1] = ld_uint4(ap0 + 8);
      af[1].q[0] = ld_uint4(ap1);
      af[1].q[1] = ld_uint4(ap1 + 8);

      Frag bfr[4];
      bfr[0].q[0] = ld_uint4(bp0);
      bfr[0].q[1] = ld_uint4(bp0 + 8);
      bfr[1].q[0] = ld_uint4(bp1);
      bfr[1].q[1] = ld_uint4(bp1 + 8);
      bfr[2].q[0] = ld_uint4(bp2);
      bfr[2].q[1] = ld_uint4(bp2 + 8);
      bfr[3].q[0] = ld_uint4(bp3);
      bfr[3].q[1] = ld_uint4(bp3 + 8);

      // Pull the next K-chunk's B blocks toward the WGP cache.
      __builtin_prefetch(bp0 + 512, 0, 3);
      __builtin_prefetch(bp1 + 512, 0, 3);
      __builtin_prefetch(bp2 + 512, 0, 3);
      __builtin_prefetch(bp3 + 512, 0, 3);

#pragma unroll
      for (int fm = 0; fm < 2; ++fm)
#pragma unroll
        for (int g = 0; g < 4; ++g)
          acc[fm][g] = __builtin_amdgcn_wmma_f32_16x16x32_bf16(
              false, af[fm].v, false, bfr[g].v, (short)0, acc[fm][g],
              false, false);

      ap0 += 32;
      ap1 += 32;
      bp0 += 512;
      bp1 += 512;
      bp2 += 512;
      bp3 += 512;
    }
  }

  // LSTM nonlinearity; D-frag layout: lane<16 -> M=r, lane>=16 -> M=8+r; N=lane&15.
#pragma unroll
  for (int fm = 0; fm < 2; ++fm) {
    const int mbase = mtile + wm * 32 + fm * 16 + ((lane >= 16) ? 8 : 0);
#pragma unroll
    for (int r = 0; r < 8; ++r) {
      const int m = mbase + r;
      const size_t idx = (size_t)m * 1024 + ncol;
      float gi = acc[fm][0][r], gf = acc[fm][1][r];
      float gg = acc[fm][2][r], go = acc[fm][3][r];
      float c_old = C[idx];
      float c_new = sigf(gf) * c_old + sigf(gi) * tanhf(gg);
      C[idx]    = c_new;
      Hout[idx] = f2bf(sigf(go) * tanhf(c_new));
    }
  }
}

// ---------------------------------------------------------------------------
// FC head: out = clip(a1 @ fcW^T + fcb, +-1).  grid=(16,6), block=32 (1 wave).
// Writes fp32 velocity to outVel[m][t][n] and (optionally) bf16 feedback x.
// ---------------------------------------------------------------------------
__global__ __launch_bounds__(32) void fc_kernel(
    const u16* __restrict__ A,        // [256 x 1024] bf16 hidden
    const u16* __restrict__ Wp,       // packed FC weights (N=96, K=1024)
    const float* __restrict__ b,      // [96]
    float* __restrict__ outVel, int t,
    u16* __restrict__ xnext) {
  const int lane = threadIdx.x & 31;
  const int m16 = blockIdx.x;   // 0..15
  const int n16 = blockIdx.y;   // 0..5
  const int ncol = n16 * 16 + (lane & 15);
  const int kbaseA = (lane >= 16) ? 16 : 0;
  float bb = b[ncol];
  v8f acc;
#pragma unroll
  for (int r = 0; r < 8; ++r) acc[r] = bb;
  const u16* ap = A + (size_t)(m16 * 16 + (lane & 15)) * 1024 + kbaseA;
  const u16* bp = Wp + ((size_t)(n16 * 32) * 32 + lane) * 16;
#pragma unroll 2
  for (int k32 = 0; k32 < 32; ++k32) {
    Frag af, bfr;
    af.q[0] = ld_uint4(ap);
    af.q[1] = ld_uint4(ap + 8);
    bfr.q[0] = ld_uint4(bp);
    bfr.q[1] = ld_uint4(bp + 8);
    __builtin_prefetch(bp + 512, 0, 3);
    acc = __builtin_amdgcn_wmma_f32_16x16x32_bf16(
        false, af.v, false, bfr.v, (short)0, acc, false, false);
    ap += 32;
    bp += 512;
  }
#pragma unroll
  for (int r = 0; r < 8; ++r) {
    const int m = m16 * 16 + r + ((lane >= 16) ? 8 : 0);
    float v = fminf(fmaxf(acc[r], -1.0f), 1.0f);
    outVel[(size_t)m * (74 * 96) + (size_t)t * 96 + ncol] = v;
    if (xnext) xnext[m * 96 + ncol] = f2bf(v);
  }
}

// ---------------------------------------------------------------------------
// Pose cumsum epilogue: out[bi][0]=pose0; out[bi][s]=pose0+cumsum(vel).
// ---------------------------------------------------------------------------
__global__ void pose_kernel(const float* __restrict__ pose,
                            const float* __restrict__ vel,
                            float* __restrict__ out, int b0, int nb) {
  int tid = blockIdx.x * blockDim.x + threadIdx.x;
  if (tid >= nb * 96) return;
  int bi = tid / 96, j = tid % 96;
  int b = b0 + bi;
  float p0 = pose[(size_t)b * 4800 + j];
  float acc = p0;
  float* o = out + (size_t)bi * (75 * 96) + j;
  o[0] = p0;
  for (int t = 0; t < 74; ++t) {
    acc += vel[(size_t)b * (74 * 96) + t * 96 + j];
    o[(size_t)(t + 1) * 96] = acc;
  }
}

// ---------------------------------------------------------------------------
// Host driver
// ---------------------------------------------------------------------------
extern "C" void kernel_launch(void* const* d_in, const int* in_sizes, int n_in,
                              void* d_out, int out_size, void* d_ws,
                              size_t ws_size, hipStream_t stream) {
  (void)in_sizes; (void)n_in; (void)out_size; (void)ws_size;

  // Input leaves in setup_inputs() insertion order (assumed):
  // 0 pose, 1 noise, then per cell {Wih,Whh,bih,bhh}:
  // enc0@2, enc1@6, dec0@10, dec1@14, dec_fcW@18, dec_fcb@19,
  // comp0@20, comp1@24, comp_fcW@28, comp_fcb@29.
  const float* pose  = (const float*)d_in[0];
  const int*   noise = (const int*)d_in[1];
  struct Cell { const float *Wih, *Whh, *bih, *bhh; };
  auto getCell = [&](int base) {
    Cell c;
    c.Wih = (const float*)d_in[base];
    c.Whh = (const float*)d_in[base + 1];
    c.bih = (const float*)d_in[base + 2];
    c.bhh = (const float*)d_in[base + 3];
    return c;
  };
  Cell cells[6] = { getCell(2), getCell(6),      // enc0, enc1
                    getCell(10), getCell(14),    // dec0, dec1
                    getCell(20), getCell(24) };  // comp0, comp1
  const int cellKx[6] = { 96, 1024, 96, 1024, 96, 1024 };
  const float* fcW[2] = { (const float*)d_in[18], (const float*)d_in[28] };
  const float* fcb[2] = { (const float*)d_in[19], (const float*)d_in[29] };

  // Workspace layout (256B aligned cursor allocator).
  char* wsb = (char*)d_ws;
  size_t cur = 0;
  auto alloc = [&](size_t bytes) -> char* {
    char* p = wsb + cur;
    cur += (bytes + 255) & ~(size_t)255;
    return p;
  };
  u16* X = (u16*)alloc(74ull * 256 * 96 * 2);            // input sequence bf16
  u16* pWx[6]; u16* pWh[6]; float* pB[6];
  for (int c = 0; c < 6; ++c) {
    pWx[c] = (u16*)alloc(4096ull * cellKx[c] * 2);
    pWh[c] = (u16*)alloc(4096ull * 1024 * 2);
    pB[c]  = (float*)alloc(4096ull * 4);
  }
  u16* pFc[2];
  for (int f = 0; f < 2; ++f) pFc[f] = (u16*)alloc(96ull * 1024 * 2);

  const size_t HB = 256ull * 1024 * 2;   // one bf16 hidden buffer
  const size_t CB = 256ull * 1024 * 4;   // one fp32 cell buffer
  // [net][layer][parity]
  u16* H[3][2][2]; float* Cst[3][2];
  for (int nt = 0; nt < 3; ++nt)
    for (int ly = 0; ly < 2; ++ly) {
      H[nt][ly][0] = (u16*)alloc(HB);
      H[nt][ly][1] = (u16*)alloc(HB);
      Cst[nt][ly]  = (float*)alloc(CB);
    }
  u16* xbuf[2] = { (u16*)alloc(256ull * 96 * 2), (u16*)alloc(256ull * 96 * 2) };

  // Output regions.
  float* out       = (float*)d_out;
  float* pred_pose = out;                          // (206,75,96)
  float* pred_vel  = pred_pose + 206ull * 75 * 96; // (256,74,96)
  float* comp_pose = pred_vel + 256ull * 74 * 96;  // (256,75,96)
  float* comp_vel  = comp_pose + 256ull * 75 * 96; // (256,74,96)

  // --- Phase 0: zero-init encoder state + decoder feedback seed -------------
  hipMemsetAsync(H[0][0][0], 0, HB, stream);
  hipMemsetAsync(H[0][1][0], 0, HB, stream);
  hipMemsetAsync(Cst[0][0], 0, CB, stream);
  hipMemsetAsync(Cst[0][1], 0, CB, stream);
  hipMemsetAsync(xbuf[0], 0, 256ull * 96 * 2, stream);

  // --- Phase 1: prep input sequence + pack weights --------------------------
  {
    int total = 74 * 256 * 96;
    prep_x_kernel<<<(total + 255) / 256, 256, 0, stream>>>(pose, noise, X);
  }
  for (int c = 0; c < 6; ++c) {
    int tx = (4096 / 16) * (cellKx[c] / 32) * 32;
    pack_w_kernel<<<(tx + 255) / 256, 256, 0, stream>>>(cells[c].Wih, pWx[c],
                                                        4096, cellKx[c]);
    int th = (4096 / 16) * (1024 / 32) * 32;
    pack_w_kernel<<<(th + 255) / 256, 256, 0, stream>>>(cells[c].Whh, pWh[c],
                                                        4096, 1024);
    bias_kernel<<<16, 256, 0, stream>>>(cells[c].bih, cells[c].bhh, pB[c], 4096);
  }
  for (int f = 0; f < 2; ++f) {
    int t = (96 / 16) * (1024 / 32) * 32;
    pack_w_kernel<<<(t + 255) / 256, 256, 0, stream>>>(fcW[f], pFc[f], 96, 1024);
  }

  const dim3 cgrid(4, 16), cblk(256);
  const dim3 fgrid(16, 6), fblk(32);

  // --- Phase 2: encoder scan ------------------------------------------------
  for (int t = 0; t < 74; ++t) {
    int p = t & 1, q = p ^ 1;
    lstm_cell_kernel<<<cgrid, cblk, 0, stream>>>(
        X + (size_t)t * 24576, 96, 96, H[0][0][p], pWx[0], pWh[0], pB[0],
        Cst[0][0], H[0][0][q]);
    lstm_cell_kernel<<<cgrid, cblk, 0, stream>>>(
        H[0][0][q], 1024, 1024, H[0][1][p], pWx[1], pWh[1], pB[1],
        Cst[0][1], H[0][1][q]);
  }
  // T=74 is even -> final encoder state lives in parity 0 buffers.

  // --- Phase 3: seed decoder (net 1) and completion (net 2) states ----------
  for (int nt = 1; nt < 3; ++nt)
    for (int ly = 0; ly < 2; ++ly) {
      hipMemcpyAsync(H[nt][ly][0], H[0][ly][0], HB, hipMemcpyDeviceToDevice,
                     stream);
      hipMemcpyAsync(Cst[nt][ly], Cst[0][ly], CB, hipMemcpyDeviceToDevice,
                     stream);
    }

  // --- Phase 4: autoregressive decoder scan ---------------------------------
  for (int t = 0; t < 74; ++t) {
    int p = t & 1, q = p ^ 1;
    lstm_cell_kernel<<<cgrid, cblk, 0, stream>>>(
        xbuf[p], 96, 96, H[1][0][p], pWx[2], pWh[2], pB[2], Cst[1][0],
        H[1][0][q]);
    lstm_cell_kernel<<<cgrid, cblk, 0, stream>>>(
        H[1][0][q], 1024, 1024, H[1][1][p], pWx[3], pWh[3], pB[3], Cst[1][1],
        H[1][1][q]);
    fc_kernel<<<fgrid, fblk, 0, stream>>>(H[1][1][q], pFc[0], fcb[0], pred_vel,
                                          t, xbuf[q]);
  }

  // --- Phase 5: completion scan (consumes X, no feedback) -------------------
  for (int t = 0; t < 74; ++t) {
    int p = t & 1, q = p ^ 1;
    lstm_cell_kernel<<<cgrid, cblk, 0, stream>>>(
        X + (size_t)t * 24576, 96, 96, H[2][0][p], pWx[4], pWh[4], pB[4],
        Cst[2][0], H[2][0][q]);
    lstm_cell_kernel<<<cgrid, cblk, 0, stream>>>(
        H[2][0][q], 1024, 1024, H[2][1][p], pWx[5], pWh[5], pB[5], Cst[2][1],
        H[2][1][q]);
    fc_kernel<<<fgrid, fblk, 0, stream>>>(H[2][1][q], pFc[1], fcb[1], comp_vel,
                                          t, (u16*)nullptr);
  }

  // --- Phase 6: pose cumsum epilogues ---------------------------------------
  pose_kernel<<<(206 * 96 + 255) / 256, 256, 0, stream>>>(pose, pred_vel,
                                                          pred_pose, 50, 206);
  pose_kernel<<<(256 * 96 + 255) / 256, 256, 0, stream>>>(pose, comp_vel,
                                                          comp_pose, 0, 256);
}
